// DualStreamGNN_80221399155597
// MI455X (gfx1250) — compile-verified, matrix operand order checked by reading, not run on hardware
//
#include <hip/hip_runtime.h>
#include <hip/hip_bf16.h>

typedef __bf16 bf16_t;
typedef bf16_t v16bf __attribute__((ext_vector_type(16)));
typedef bf16_t v8bf  __attribute__((ext_vector_type(8)));
typedef float  v8f   __attribute__((ext_vector_type(8)));
typedef int    v4i_g __attribute__((vector_size(4 * sizeof(int))));

#define B_SZ    512
#define T_SZ    1000
#define IN_DIMC 32
#define TEACH_D 26
#define LSTM_H  128
#define OUT_D   26
#define KPAD    224      // LSTM_IN 208 padded to 7*32
#define NG      512      // 4*LSTM_H
#define BSLICE  32       // batch rows per WG (2 M-tiles)
#define NWG     (B_SZ / BSLICE)   // 16 workgroups
#define HSTR    136      // padded bf16 row stride for h/Whh0 (272B)
#define SEQSTR  232      // padded bf16 row stride for LDS seq tile (464B, conflict-free)

#define SEQ_ROW_B   (KPAD * 2)            // 448B per global seq row
#define SEQ_LROW_B  (SEQSTR * 2)          // 464B per LDS seq row
#define SEQ_CHUNKS  (BSLICE * (SEQ_ROW_B / 16))   // 896 x 16B chunks per tile

// dynamic LDS layout
#define SMEM_WHH0_B  ((size_t)NG * HSTR * 2)               // 139264
#define SMEM_H_B     ((size_t)BSLICE * HSTR * 2)           // 8704
#define SMEM_SEQ_B   ((size_t)BSLICE * SEQ_LROW_B)         // 14848
#define SMEM_BYTES   (SMEM_WHH0_B + 2 * SMEM_H_B + 2 * SMEM_SEQ_B)  // 186368

#define AS1 __attribute__((address_space(1)))
#define AS3 __attribute__((address_space(3)))

#if defined(__HIP_DEVICE_COMPILE__) && __has_builtin(__builtin_amdgcn_global_load_async_to_lds_b128) && __has_builtin(__builtin_amdgcn_s_wait_asynccnt)
#define HAVE_ASYNC_LDS 1
#else
#define HAVE_ASYNC_LDS 0
#endif

// ---------------------------------------------------------------------------
__device__ __forceinline__ v8f wmma_bf16(v16bf a, v16bf b, v8f c) {
  return __builtin_amdgcn_wmma_f32_16x16x32_bf16(false, a, false, b, (short)0, c,
                                                 false, false);
}

// 16(row) x 32(K) fragment loader (A: row=M; B: weights row-major [N][K]).
__device__ __forceinline__ v16bf load_frag(const bf16_t* base, int ld, int k0) {
  const int lane = threadIdx.x & 31;
  const int r    = lane & 15;
  const int kb   = (lane >> 4) * 8;
  const bf16_t* p = base + (size_t)r * ld + k0 + kb;
  v8bf lo = *(const v8bf*)(p);
  v8bf hi = *(const v8bf*)(p + 16);
  v16bf f;
#pragma unroll
  for (int i = 0; i < 8; ++i) { f[i] = lo[i]; f[8 + i] = hi[i]; }
  return f;
}

__device__ __forceinline__ float sigmoidf(float x) {
  return 1.0f / (1.0f + __expf(-x));
}

// Stage one 32xKPAD bf16 seq tile into LDS (restrided). Async when available.
__device__ __forceinline__ void stage_seq_tile(const bf16_t* gbase, char* lbase) {
  for (int c = threadIdx.x; c < SEQ_CHUNKS; c += 256) {
    const int row = c / (SEQ_ROW_B / 16);
    const int col = c - row * (SEQ_ROW_B / 16);
    const char* g = (const char*)gbase + row * SEQ_ROW_B + col * 16;
    char* l = lbase + row * SEQ_LROW_B + col * 16;
#if HAVE_ASYNC_LDS
    __builtin_amdgcn_global_load_async_to_lds_b128(
        (AS1 v4i_g*)(unsigned long long)(const void*)g,
        (AS3 v4i_g*)(unsigned)(unsigned long long)(void*)l, 0, 0);
#else
    *(int4*)l = *(const int4*)g;
#endif
  }
}

__device__ __forceinline__ void wait_seq_tile() {
#if HAVE_ASYNC_LDS
  __builtin_amdgcn_s_wait_asynccnt(0);
#endif
}

// ---------------------------------------------------------------------------
// Kernel 1: weights -> bf16 (+K padding), bias folding
// ---------------------------------------------------------------------------
__global__ void convert_kernel(const float* __restrict__ Wih0, const float* __restrict__ Whh0,
                               const float* __restrict__ bih0, const float* __restrict__ bhh0,
                               const float* __restrict__ Wih1, const float* __restrict__ Whh1,
                               const float* __restrict__ bih1, const float* __restrict__ bhh1,
                               bf16_t* __restrict__ wih0b, bf16_t* __restrict__ whh0b,
                               bf16_t* __restrict__ wih1b, bf16_t* __restrict__ whh1b,
                               float* __restrict__ b0o, float* __restrict__ b1o) {
  const int i = blockIdx.x * blockDim.x + threadIdx.x;
  if (i < NG * KPAD) {
    const int n = i / KPAD, k = i - n * KPAD;
    wih0b[i] = (bf16_t)((k < 208) ? Wih0[n * 208 + k] : 0.0f);
  }
  if (i < NG * LSTM_H) {
    whh0b[i] = (bf16_t)Whh0[i];
    wih1b[i] = (bf16_t)Wih1[i];
    whh1b[i] = (bf16_t)Whh1[i];
  }
  if (i < NG) {
    b0o[i] = bih0[i] + bhh0[i];
    b1o[i] = bih1[i] + bhh1[i];
  }
}

// ---------------------------------------------------------------------------
// Kernel 2: seq[T][B][KPAD] bf16 = [g | g | pe | te | zero-pad]
// ---------------------------------------------------------------------------
__global__ __launch_bounds__(64) void preprocess_kernel(
    const float* __restrict__ xs, const float* __restrict__ xt,
    const float* __restrict__ Wg, const float* __restrict__ bg,
    const float* __restrict__ Wt, const float* __restrict__ bt,
    bf16_t* __restrict__ seq) {
  const int idx = blockIdx.x;  // b * T + t
  const int b = idx / T_SZ;
  const int t = idx - b * T_SZ;
  const int j = threadIdx.x;   // 0..63
  __shared__ float mavg[16];
  __shared__ float xv[TEACH_D];
  if (j < 16) {
    const size_t o = ((size_t)b * T_SZ + t) * IN_DIMC;
    mavg[j] = 0.5f * (xs[o + j] + xs[o + 16 + j]);
  }
  if (j < TEACH_D) xv[j] = xt[((size_t)b * T_SZ + t) * TEACH_D + j];
  __syncthreads();
  float g = bg[j];
#pragma unroll
  for (int k = 0; k < 16; ++k) g += mavg[k] * Wg[k * 64 + j];
  float te = bt[j];
#pragma unroll
  for (int k = 0; k < TEACH_D; ++k) te += xv[k] * Wt[k * 64 + j];
  bf16_t* o = seq + ((size_t)t * B_SZ + b) * KPAD;
  o[j]       = (bf16_t)g;
  o[64 + j]  = (bf16_t)g;
  o[144 + j] = (bf16_t)te;
  if (j < 16) {
    const int i2 = (j >> 1) << 1;
    const float div = __expf(-(float)i2 * (logf(10000.0f) / 16.0f));
    const float ang = (float)t * div;
    o[128 + j] = (bf16_t)((j & 1) ? cosf(ang) : sinf(ang));
    o[208 + j] = (bf16_t)0.0f;
  }
}

// ---------------------------------------------------------------------------
// Kernel 3: fused 2-layer LSTM, persistent over T. One WG per 32-batch slice.
// Wave w owns hidden units [16w,16w+16) across all four gates -> cell update
// runs on WMMA accumulators, c-state in VGPRs. h-state + Whh0 + double-
// buffered seq tile in LDS (seq staged via async-to-LDS, ASYNCcnt-tracked).
// Wih0/Wih1/Whh1 stream from L2 (laundered against LICM hoist+spill).
// ---------------------------------------------------------------------------
__global__ __launch_bounds__(256) void lstm_fused_kernel(
    const bf16_t* __restrict__ seq,   // [T][B][KPAD]
    const bf16_t* __restrict__ Wih0,  // [NG][KPAD]
    const bf16_t* __restrict__ Whh0,  // [NG][LSTM_H]
    const bf16_t* __restrict__ Wih1,  // [NG][LSTM_H]
    const bf16_t* __restrict__ Whh1,  // [NG][LSTM_H]
    const float* __restrict__ bias0,  // [NG]
    const float* __restrict__ bias1,  // [NG]
    float* __restrict__ h_final) {    // [B][LSTM_H]
  extern __shared__ char smem[];
  bf16_t* whh0s = (bf16_t*)smem;                               // [NG][HSTR]
  bf16_t* h0s   = (bf16_t*)(smem + SMEM_WHH0_B);               // [BSLICE][HSTR]
  bf16_t* h1s   = (bf16_t*)(smem + SMEM_WHH0_B + SMEM_H_B);    // [BSLICE][HSTR]
  char*   seqb0 = smem + SMEM_WHH0_B + 2 * SMEM_H_B;           // [BSLICE][SEQSTR]
  char*   seqb1 = seqb0 + SMEM_SEQ_B;

  const int tid  = threadIdx.x;
  const int wave = tid >> 5;
  const int lane = tid & 31;
  const int b0   = blockIdx.x * BSLICE;
  const int j    = wave * 16 + (lane & 15);   // owned hidden unit (D column)
  const int mb   = (lane >> 4) * 8;           // accumulator row base in tile

  // Stage Whh0 into LDS with bank-spreading stride.
  for (int r = tid; r < NG; r += 256) {
    const int4* src = (const int4*)(Whh0 + (size_t)r * LSTM_H);
    int4* dst = (int4*)(whh0s + (size_t)r * HSTR);
#pragma unroll
    for (int c4 = 0; c4 < 8; ++c4) dst[c4] = src[c4];
  }
  for (int i = tid; i < BSLICE * HSTR; i += 256) {
    h0s[i] = (bf16_t)0.0f;
    h1s[i] = (bf16_t)0.0f;
  }

  const float bi0 = bias0[j], bf0 = bias0[128 + j], bg0 = bias0[256 + j], bo0 = bias0[384 + j];
  const float bi1 = bias1[j], bf1 = bias1[128 + j], bg1 = bias1[256 + j], bo1 = bias1[384 + j];

  v8f c0[2] = {v8f{}, v8f{}};
  v8f c1[2] = {v8f{}, v8f{}};

  // Kick off the first seq tile before anyone needs it.
  stage_seq_tile(seq + (size_t)b0 * KPAD, seqb0);
  __syncthreads();

  for (int t = 0; t < T_SZ; ++t) {
    // Opaque zero: keeps weight fragment loads inside the time loop.
    int lz = 0;
    asm volatile("" : "+v"(lz));
    const bf16_t* wih0p  = Wih0 + lz;
    const bf16_t* wih1p  = Wih1 + lz;
    const bf16_t* whh1p  = Whh1 + lz;
    const bf16_t* whh0sp = whh0s + lz;

    const bf16_t* seqcur = (const bf16_t*)((t & 1) ? seqb1 : seqb0);
    char* seqnxt = (t & 1) ? seqb0 : seqb1;

    wait_seq_tile();   // this wave's async chunks of tile t have landed
    __syncthreads();   // all waves' chunks have landed
    if (t + 1 < T_SZ) {
      stage_seq_tile(seq + ((size_t)(t + 1) * B_SZ + b0) * KPAD, seqnxt);
      __builtin_prefetch(seq + ((size_t)(t + 1) * B_SZ + b0) * KPAD + (size_t)tid * 56, 0, 1);
    }

    // ---------------- layer 0: z = seq_t @ Wih0^T + h0 @ Whh0^T ----------
    v8f acc[2][4];
#pragma unroll
    for (int mt = 0; mt < 2; ++mt)
#pragma unroll
      for (int g = 0; g < 4; ++g) acc[mt][g] = v8f{};

#pragma unroll
    for (int kc = 0; kc < KPAD / 32; ++kc) {
      const v16bf a0 = load_frag(seqcur, SEQSTR, kc * 32);
      const v16bf a1 = load_frag(seqcur + (size_t)16 * SEQSTR, SEQSTR, kc * 32);
#pragma unroll
      for (int g = 0; g < 4; ++g) {
        const v16bf bb = load_frag(wih0p + (size_t)(g * 128 + wave * 16) * KPAD, KPAD, kc * 32);
        acc[0][g] = wmma_bf16(a0, bb, acc[0][g]);
        acc[1][g] = wmma_bf16(a1, bb, acc[1][g]);
      }
    }
#pragma unroll
    for (int kc = 0; kc < LSTM_H / 32; ++kc) {
      const v16bf a0 = load_frag(h0s, HSTR, kc * 32);
      const v16bf a1 = load_frag(h0s + (size_t)16 * HSTR, HSTR, kc * 32);
#pragma unroll
      for (int g = 0; g < 4; ++g) {
        const v16bf bb = load_frag(whh0sp + (size_t)(g * 128 + wave * 16) * HSTR, HSTR, kc * 32);
        acc[0][g] = wmma_bf16(a0, bb, acc[0][g]);
        acc[1][g] = wmma_bf16(a1, bb, acc[1][g]);
      }
    }
    __syncthreads();   // everyone finished reading h0(t-1)

#pragma unroll
    for (int mt = 0; mt < 2; ++mt) {
#pragma unroll
      for (int r = 0; r < 8; ++r) {
        const float zi = acc[mt][0][r] + bi0;
        const float zf = acc[mt][1][r] + bf0;
        const float zg = acc[mt][2][r] + bg0;
        const float zo = acc[mt][3][r] + bo0;
        const float cn = sigmoidf(zf) * c0[mt][r] + sigmoidf(zi) * tanhf(zg);
        c0[mt][r] = cn;
        const float hn = sigmoidf(zo) * tanhf(cn);
        h0s[(mt * 16 + mb + r) * HSTR + j] = (bf16_t)hn;
      }
    }
    __syncthreads();   // h0(t) visible

    // ---------------- layer 1: z = h0 @ Wih1^T + h1 @ Whh1^T -------------
#pragma unroll
    for (int mt = 0; mt < 2; ++mt)
#pragma unroll
      for (int g = 0; g < 4; ++g) acc[mt][g] = v8f{};

#pragma unroll
    for (int kc = 0; kc < LSTM_H / 32; ++kc) {
      const v16bf a0 = load_frag(h0s, HSTR, kc * 32);
      const v16bf a1 = load_frag(h0s + (size_t)16 * HSTR, HSTR, kc * 32);
#pragma unroll
      for (int g = 0; g < 4; ++g) {
        const v16bf bb = load_frag(wih1p + (size_t)(g * 128 + wave * 16) * LSTM_H, LSTM_H, kc * 32);
        acc[0][g] = wmma_bf16(a0, bb, acc[0][g]);
        acc[1][g] = wmma_bf16(a1, bb, acc[1][g]);
      }
    }
#pragma unroll
    for (int kc = 0; kc < LSTM_H / 32; ++kc) {
      const v16bf a0 = load_frag(h1s, HSTR, kc * 32);
      const v16bf a1 = load_frag(h1s + (size_t)16 * HSTR, HSTR, kc * 32);
#pragma unroll
      for (int g = 0; g < 4; ++g) {
        const v16bf bb = load_frag(whh1p + (size_t)(g * 128 + wave * 16) * LSTM_H, LSTM_H, kc * 32);
        acc[0][g] = wmma_bf16(a0, bb, acc[0][g]);
        acc[1][g] = wmma_bf16(a1, bb, acc[1][g]);
      }
    }
    __syncthreads();   // everyone finished reading h1(t-1)

#pragma unroll
    for (int mt = 0; mt < 2; ++mt) {
#pragma unroll
      for (int r = 0; r < 8; ++r) {
        const float zi = acc[mt][0][r] + bi1;
        const float zf = acc[mt][1][r] + bf1;
        const float zg = acc[mt][2][r] + bg1;
        const float zo = acc[mt][3][r] + bo1;
        const float cn = sigmoidf(zf) * c1[mt][r] + sigmoidf(zi) * tanhf(zg);
        c1[mt][r] = cn;
        const float hn = sigmoidf(zo) * tanhf(cn);
        h1s[(mt * 16 + mb + r) * HSTR + j] = (bf16_t)hn;
        if (t == T_SZ - 1) {
          h_final[(size_t)(b0 + mt * 16 + mb + r) * LSTM_H + j] = hn;
        }
      }
    }
    __syncthreads();
  }
}

// ---------------------------------------------------------------------------
// Kernel 4: MLP head 128 -> relu 128 -> relu 64 -> 26 (tiny, plain f32)
// ---------------------------------------------------------------------------
__global__ __launch_bounds__(128) void head_kernel(
    const float* __restrict__ hin, const float* __restrict__ W1, const float* __restrict__ b1,
    const float* __restrict__ W2, const float* __restrict__ b2,
    const float* __restrict__ W3, const float* __restrict__ b3,
    float* __restrict__ out) {
  const int b = blockIdx.x;
  const int j = threadIdx.x;
  __shared__ float h[128], y1[128], y2[64];
  h[j] = hin[(size_t)b * 128 + j];
  __syncthreads();
  float a = b1[j];
#pragma unroll 8
  for (int k = 0; k < 128; ++k) a += h[k] * W1[j * 128 + k];
  y1[j] = fmaxf(a, 0.0f);
  __syncthreads();
  if (j < 64) {
    float a2 = b2[j];
#pragma unroll 8
    for (int k = 0; k < 128; ++k) a2 += y1[k] * W2[j * 128 + k];
    y2[j] = fmaxf(a2, 0.0f);
  }
  __syncthreads();
  if (j < OUT_D) {
    float a3 = b3[j];
#pragma unroll 8
    for (int k = 0; k < 64; ++k) a3 += y2[k] * W3[j * 64 + k];
    out[(size_t)b * OUT_D + j] = a3;
  }
}

// ---------------------------------------------------------------------------
extern "C" void kernel_launch(void* const* d_in, const int* in_sizes, int n_in,
                              void* d_out, int out_size, void* d_ws, size_t ws_size,
                              hipStream_t stream) {
  const float* xs   = (const float*)d_in[0];
  const float* xt   = (const float*)d_in[1];
  const float* Wg   = (const float*)d_in[2];
  const float* bg   = (const float*)d_in[3];
  const float* Wt   = (const float*)d_in[4];
  const float* bt   = (const float*)d_in[5];
  const float* Wih0 = (const float*)d_in[6];
  const float* Whh0 = (const float*)d_in[7];
  const float* bih0 = (const float*)d_in[8];
  const float* bhh0 = (const float*)d_in[9];
  const float* Wih1 = (const float*)d_in[10];
  const float* Whh1 = (const float*)d_in[11];
  const float* bih1 = (const float*)d_in[12];
  const float* bhh1 = (const float*)d_in[13];
  const float* W1   = (const float*)d_in[14];
  const float* b1   = (const float*)d_in[15];
  const float* W2   = (const float*)d_in[16];
  const float* b2   = (const float*)d_in[17];
  const float* W3   = (const float*)d_in[18];
  const float* b3   = (const float*)d_in[19];
  float* out = (float*)d_out;

  char* ws = (char*)d_ws;
  size_t off = 0;
  auto carve = [&](size_t bytes) {
    char* p = ws + off;
    off = (off + bytes + 255) & ~(size_t)255;
    return p;
  };
  bf16_t* seqb  = (bf16_t*)carve((size_t)T_SZ * B_SZ * KPAD * 2);  // 229 MB
  bf16_t* wih0b = (bf16_t*)carve((size_t)NG * KPAD * 2);
  bf16_t* whh0b = (bf16_t*)carve((size_t)NG * LSTM_H * 2);
  bf16_t* wih1b = (bf16_t*)carve((size_t)NG * LSTM_H * 2);
  bf16_t* whh1b = (bf16_t*)carve((size_t)NG * LSTM_H * 2);
  float*  b0f   = (float*)carve((size_t)NG * 4);
  float*  b1f   = (float*)carve((size_t)NG * 4);
  float*  hfin  = (float*)carve((size_t)B_SZ * LSTM_H * 4);
  (void)ws_size; (void)in_sizes; (void)n_in; (void)out_size;

  convert_kernel<<<(NG * KPAD + 255) / 256, 256, 0, stream>>>(
      Wih0, Whh0, bih0, bhh0, Wih1, Whh1, bih1, bhh1,
      wih0b, whh0b, wih1b, whh1b, b0f, b1f);

  preprocess_kernel<<<B_SZ * T_SZ, 64, 0, stream>>>(xs, xt, Wg, bg, Wt, bt, seqb);

  lstm_fused_kernel<<<NWG, 256, SMEM_BYTES, stream>>>(seqb, wih0b, whh0b, wih1b,
                                                      whh1b, b0f, b1f, hfin);

  head_kernel<<<B_SZ, 128, 0, stream>>>(hfin, W1, b1, W2, b2, W3, b3, out);
}
